// QuantumLSTMLayer_1675037245564
// MI455X (gfx1250) — compile-verified, matrix operand order checked by reading, not run on hardware
//
#include <hip/hip_runtime.h>
#include <hip/hip_bf16.h>
#include <math.h>

typedef __attribute__((ext_vector_type(16))) _Float16 v16h;
typedef __attribute__((ext_vector_type(8)))  float    v8f;

#define NQ    10
#define DIM   1024
#define BSZ   2048
#define TT    8
#define PW    81        // 8*N+1
#define FEATW 30        // 3*N

#if defined(__gfx1250__) && __has_builtin(__builtin_amdgcn_tensor_load_to_lds)
#  define HAVE_TDM 1
typedef unsigned int u32x4 __attribute__((ext_vector_type(4)));
typedef int          i32x8 __attribute__((ext_vector_type(8)));
typedef int          i32x4 __attribute__((ext_vector_type(4)));
#else
#  define HAVE_TDM 0
#endif

// ---- CZ pair masks (qubit q <-> bit (9-q) of the amplitude index) ----
__constant__ unsigned EVEN_CZ_MASKS[5] = {
    (1u<<9)|(1u<<8), (1u<<7)|(1u<<6), (1u<<5)|(1u<<4), (1u<<3)|(1u<<2), (1u<<1)|(1u<<0)
};
// CAND_CZ from np.random.RandomState(0): pairs
// (0,2)(0,8)(0,9)(1,3)(1,6)(2,3)(2,4)(2,5)(2,6)(2,7)(2,9)(3,7)(4,6)(5,9)
__constant__ unsigned CAND_CZ_MASKS[14] = {
    (1u<<9)|(1u<<7), (1u<<9)|(1u<<1), (1u<<9)|(1u<<0),
    (1u<<8)|(1u<<6), (1u<<8)|(1u<<3),
    (1u<<7)|(1u<<6), (1u<<7)|(1u<<5), (1u<<7)|(1u<<4), (1u<<7)|(1u<<3), (1u<<7)|(1u<<2), (1u<<7)|(1u<<0),
    (1u<<6)|(1u<<2), (1u<<5)|(1u<<3), (1u<<4)|(1u<<0)
};

// ---------------------------------------------------------------------
__global__ void zero_kernel(float* buf, int n) {
    int i = blockIdx.x * blockDim.x + threadIdx.x;
    if (i < n) buf[i] = 0.f;
}

// ---------------------------------------------------------------------
// GEMM1: p[b,j] = sum_k x[b,t,k]*Wx[k,j] + sum_k h[b,k]*Wh[k,j] + bias[j]
// One wave per 16x16 tile; K=20 packed into a single 16x16x32 f16 WMMA.
// Branchless clamped gathers -> v_cndmask, no exec-mask churn.
// grid (128, 6), block 32.
// ---------------------------------------------------------------------
__global__ void gemm_p_kernel(const float* __restrict__ x, int t,
                              const float* __restrict__ Wx,
                              const float* __restrict__ Wh,
                              const float* __restrict__ bias,
                              const float* __restrict__ h,
                              float* __restrict__ p) {
    const int tileM = blockIdx.x, tileJ = blockIdx.y;
    const int lane  = threadIdx.x & 31;
    const int half  = lane >> 4;
    const int M     = tileM * 16 + (lane & 15);       // batch row for A
    const int Jc    = tileJ * 16 + (lane & 15);       // output col for B
    const int JcC   = (Jc < PW) ? Jc : 0;             // clamped, always safe
    const float jm  = (Jc < PW) ? 1.f : 0.f;

    __builtin_prefetch(Wx + JcC, 0, 0);               // global_prefetch_b8
    __builtin_prefetch(Wh + JcC, 0, 0);

    v16h a, bm;
#pragma unroll
    for (int e = 0; e < 16; ++e) {
        const int v  = e >> 1, hi = e & 1;
        // CDNA5 16-bit A layout: lanes 0-15 hold K=0..7 (v0-3), 16..23 (v4-7);
        // lanes 16-31 hold K=8..15 and 24..31.
        const int k = (v < 4) ? (8 * half + 2 * v + hi)
                              : (16 + 8 * half + 2 * (v - 4) + hi);
        const int  kx = (k < 10) ? k : 0;                       // safe idx
        const int  kh = (k >= 10 && k < 20) ? (k - 10) : 0;     // safe idx
        const float xv  = x[(M * TT + t) * NQ + kx];
        const float hv  = h[M * NQ + kh];
        const float wxv = Wx[kx * PW + JcC];
        const float whv = Wh[kh * PW + JcC];
        const float av = (k < 10) ? xv : ((k < 20) ? hv : 0.f);
        const float bv = ((k < 10) ? wxv : ((k < 20) ? whv : 0.f)) * jm;
        a[e]  = (_Float16)av;
        bm[e] = (_Float16)bv;
    }
    v8f acc = {};
    acc = __builtin_amdgcn_wmma_f32_16x16x32_f16(false, a, false, bm,
                                                 (short)0, acc, false, false);
    if (Jc < PW) {
        const float bb = bias[Jc];
#pragma unroll
        for (int r = 0; r < 8; ++r) {       // D: VGPR r -> M = r + 8*half
            const int row = tileM * 16 + r + 8 * half;
            p[row * PW + Jc] = acc[r] + bb;
        }
    }
}

// ---------------------------------------------------------------------
// Circuit simulation helpers (state in LDS: re[1024], im[1024])
// ---------------------------------------------------------------------
__device__ __forceinline__ void apply_gate(float* sre, float* sim_, int q,
    float u00r, float u00i, float u01r, float u01i,
    float u10r, float u10i, float u11r, float u11i, int tid) {
    const int bt = 9 - q;
    for (int pr = tid; pr < 512; pr += 256) {
        const int low = pr & ((1 << bt) - 1);
        const int i0  = ((pr >> bt) << (bt + 1)) | low;
        const int i1  = i0 | (1 << bt);
        const float a0r = sre[i0], a0i = sim_[i0];
        const float a1r = sre[i1], a1i = sim_[i1];
        sre[i0]  = u00r*a0r - u00i*a0i + u01r*a1r - u01i*a1i;
        sim_[i0] = u00r*a0i + u00i*a0r + u01r*a1i + u01i*a1r;
        sre[i1]  = u10r*a0r - u10i*a0i + u11r*a1r - u11i*a1i;
        sim_[i1] = u10r*a0i + u10i*a0r + u11r*a1i + u11i*a1r;
    }
    __syncthreads();
}

__device__ __forceinline__ void apply_diag(float* sre, float* sim_, int q,
    float d0r, float d0i, float d1r, float d1i, int tid) {
    const int bt = 9 - q;
    for (int i = tid; i < DIM; i += 256) {
        const bool one = (i >> bt) & 1;
        const float dr = one ? d1r : d0r;
        const float di = one ? d1i : d0i;
        const float r_ = sre[i], m_ = sim_[i];
        sre[i]  = r_ * dr - m_ * di;
        sim_[i] = r_ * di + m_ * dr;
    }
    __syncthreads();
}

__device__ __forceinline__ void apply_cnot(float* sre, float* sim_, int qc, int qt, int tid) {
    const int bc = 9 - qc, bt = 9 - qt;
    for (int pr = tid; pr < 512; pr += 256) {
        const int low = pr & ((1 << bt) - 1);
        const int i0  = ((pr >> bt) << (bt + 1)) | low;
        const int i1  = i0 | (1 << bt);
        if ((i0 >> bc) & 1) {
            float tr = sre[i0];  sre[i0]  = sre[i1];  sre[i1]  = tr;
            float ti = sim_[i0]; sim_[i0] = sim_[i1]; sim_[i1] = ti;
        }
    }
    __syncthreads();
}

__device__ __forceinline__ void apply_cz_masks(float* sre, float* sim_,
                                               const unsigned* masks, int nm, int tid) {
    for (int i = tid; i < DIM; i += 256) {
        int par = 0;
        for (int k = 0; k < nm; ++k)
            par ^= (((unsigned)i & masks[k]) == masks[k]) ? 1 : 0;
        if (par) { sre[i] = -sre[i]; sim_[i] = -sim_[i]; }
    }
    __syncthreads();
}

__device__ __forceinline__ void measure_all(const float* sre, const float* sim_,
                                            float* feats_row, float* acc, int tid) {
    for (int q = 0; q < NQ; ++q) {
        const int bt = 9 - q;
        float d = 0.f, cr = 0.f, ci = 0.f;
        for (int pr = tid; pr < 512; pr += 256) {
            const int low = pr & ((1 << bt) - 1);
            const int i0  = ((pr >> bt) << (bt + 1)) | low;
            const int i1  = i0 | (1 << bt);
            const float a0r = sre[i0], a0i = sim_[i0];
            const float a1r = sre[i1], a1i = sim_[i1];
            d  += a0r*a0r + a0i*a0i - a1r*a1r - a1i*a1i;
            cr += a0r*a1r + a0i*a1i;       // Re(conj(a0)*a1)
            ci += a0r*a1i - a0i*a1r;       // Im(conj(a0)*a1)
        }
        // wave32 reduction
#pragma unroll
        for (int off = 16; off > 0; off >>= 1) {
            d  += __shfl_xor(d,  off, 32);
            cr += __shfl_xor(cr, off, 32);
            ci += __shfl_xor(ci, off, 32);
        }
        if (tid < 3) acc[tid] = 0.f;
        __syncthreads();
        if ((tid & 31) == 0) {              // one lane per wave -> ds_add_f32
            atomicAdd(&acc[0], d);
            atomicAdd(&acc[1], cr);
            atomicAdd(&acc[2], ci);
        }
        __syncthreads();
        if (tid == 0) {
            feats_row[3*q + 0] = acc[0];
            feats_row[3*q + 1] = 2.f * acc[1];
            feats_row[3*q + 2] = 2.f * acc[2];
        }
        __syncthreads();
    }
}

// ---------------------------------------------------------------------
// One block = one (batch element, circuit). grid (2048, 6), block 256.
// p row (81 floats) staged into LDS once via the Tensor Data Mover.
// feats layout: [circ][b][30]
// ---------------------------------------------------------------------
__global__ void circuits_kernel(const float* __restrict__ p,
                                float* __restrict__ feats) {
    __shared__ float sre[DIM];
    __shared__ float sim_[DIM];
    __shared__ float spb[PW];      // staged gate-parameter row
    __shared__ float acc[3];
    const int tid  = threadIdx.x;
    const int b    = blockIdx.x;
    const int circ = blockIdx.y;

#if HAVE_TDM
    // TDM: DMA p[b, 0:81] (324 B, 1-D tile) into LDS; wave 0 issues it.
    if (tid < 32) {
        const unsigned long long ga =
            (unsigned long long)(const void*)(p + (size_t)b * PW);
        const unsigned lds_off = (unsigned)(unsigned long long)(const void*)spb;
        u32x4 g0;
        g0[0] = 1u;                                            // count=1 (user D#)
        g0[1] = lds_off;                                       // lds_addr
        g0[2] = (unsigned)(ga & 0xFFFFFFFFull);                // global_addr[31:0]
        g0[3] = (unsigned)((ga >> 32) & 0x1FFFFFFull)          // global_addr[56:32]
              | (2u << 30);                                    // type=2 ("image")
        i32x8 g1;
        g1[0] = (int)(2u << 16);       // workgroup_mask=0, data_size=4B
        g1[1] = (int)((unsigned)PW << 16);  // tensor_dim0 = 81 (bits 63:48)
        g1[2] = (int)(1u << 16);       // tensor_dim0 hi = 0, tensor_dim1 = 1
        g1[3] = (int)((unsigned)PW << 16);  // tensor_dim1 hi = 0, tile_dim0 = 81
        g1[4] = 1;                     // tile_dim1 = 1, tile_dim2 = 0
        g1[5] = PW;                    // tensor_dim0_stride = 81
        g1[6] = (int)((unsigned)PW << 16);  // stride hi = 0, tensor_dim1_stride lo
        g1[7] = 0;
        const i32x4 z4 = {0, 0, 0, 0};                 // groups 2/3 unused (<=2D)
        const i32x8 z8 = {0, 0, 0, 0, 0, 0, 0, 0};     // extra group (clang-23 form)
        __builtin_amdgcn_tensor_load_to_lds(g0, g1, z4, z4, z8, 0);
        __builtin_amdgcn_s_wait_tensorcnt(0);                  // s_wait_tensorcnt
    }
#else
    for (int i = tid; i < PW; i += 256) spb[i] = p[b * PW + i];
#endif
    for (int i = tid; i < DIM; i += 256) { sre[i] = 0.03125f; sim_[i] = 0.f; }
    __syncthreads();

    const float* pb = spb;          // gate params now come from LDS
    const float R2 = 0.70710678118654752f;

    switch (circ) {
    case 0: { // forget: RY(pf[q]) ; CZ even ; T on all qubits
        for (int q = 0; q < NQ; ++q) {
            const float th = 0.5f * pb[q];
            const float c = __cosf(th), s = __sinf(th);
            apply_gate(sre, sim_, q, c,0.f, -s,0.f, s,0.f, c,0.f, tid);
        }
        apply_cz_masks(sre, sim_, EVEN_CZ_MASKS, 5, tid);
        for (int i = tid; i < DIM; i += 256) {   // fused T^popcount phase
            const float ang = 0.78539816339744831f * (float)__popc(i);
            const float cc = __cosf(ang), ss = __sinf(ang);
            const float r_ = sre[i], m_ = sim_[i];
            sre[i]  = r_ * cc - m_ * ss;
            sim_[i] = r_ * ss + m_ * cc;
        }
        __syncthreads();
    } break;
    case 1: { // input: RY,RZ per qubit ; CNOT chain
        for (int q = 0; q < NQ; ++q) {
            float th = 0.5f * pb[10 + 2*q];
            float c = __cosf(th), s = __sinf(th);
            apply_gate(sre, sim_, q, c,0.f, -s,0.f, s,0.f, c,0.f, tid);
            th = 0.5f * pb[11 + 2*q];
            c = __cosf(th); s = __sinf(th);
            apply_diag(sre, sim_, q, c,-s, c,s, tid);   // RZ
        }
        for (int q = 0; q < NQ - 1; ++q) apply_cnot(sre, sim_, q, q + 1, tid);
    } break;
    case 2: { // output: RX per qubit ; CNOT chain
        for (int q = 0; q < NQ; ++q) {
            const float th = 0.5f * pb[30 + q];
            const float c = __cosf(th), s = __sinf(th);
            apply_gate(sre, sim_, q, c,0.f, 0.f,-s, 0.f,-s, c,0.f, tid);
        }
        for (int q = 0; q < NQ - 1; ++q) apply_cnot(sre, sim_, q, q + 1, tid);
    } break;
    case 3: { // candidate: H all ; RZ,RY,RZ per qubit ; CAND_CZ
        for (int q = 0; q < NQ; ++q)
            apply_gate(sre, sim_, q, R2,0.f, R2,0.f, R2,0.f, -R2,0.f, tid);
        for (int q = 0; q < NQ; ++q) {
            float th = 0.5f * pb[40 + 3*q];
            float c = __cosf(th), s = __sinf(th);
            apply_diag(sre, sim_, q, c,-s, c,s, tid);
            th = 0.5f * pb[41 + 3*q];
            c = __cosf(th); s = __sinf(th);
            apply_gate(sre, sim_, q, c,0.f, -s,0.f, s,0.f, c,0.f, tid);
            th = 0.5f * pb[42 + 3*q];
            c = __cosf(th); s = __sinf(th);
            apply_diag(sre, sim_, q, c,-s, c,s, tid);
        }
        apply_cz_masks(sre, sim_, CAND_CZ_MASKS, 14, tid);
    } break;
    case 4: { // memory: RY per qubit ; CZ even
        for (int q = 0; q < NQ; ++q) {
            const float th = 0.5f * pb[70 + q];
            const float c = __cosf(th), s = __sinf(th);
            apply_gate(sre, sim_, q, c,0.f, -s,0.f, s,0.f, c,0.f, tid);
        }
        apply_cz_masks(sre, sim_, EVEN_CZ_MASKS, 5, tid);
    } break;
    default: { // temporal
        const float tv = pb[80];
        for (int i = 0; i < 5; ++i) apply_cnot(sre, sim_, i, 5 + i, tid);
        apply_gate(sre, sim_, 0, R2,0.f, R2,0.f, R2,0.f, -R2,0.f, tid);
        for (int i = 1; i < NQ; ++i) apply_cnot(sre, sim_, 0, i, tid);
        {
            const float thz = 0.5f * 0.1f  * tv;
            const float cz_ = __cosf(thz), sz_ = __sinf(thz);
            const float thx = 0.5f * 0.05f * tv;
            const float cx_ = __cosf(thx), sx_ = __sinf(thx);
            for (int q = 0; q < NQ; ++q) {
                apply_diag(sre, sim_, q, cz_,-sz_, cz_,sz_, tid);          // RZ(0.1t)
                apply_gate(sre, sim_, q, cx_,0.f, 0.f,-sx_, 0.f,-sx_, cx_,0.f, tid); // RX(0.05t)
            }
        }
        {
            const float th2 = 0.5f * 0.2f * tv;
            const float c2 = __cosf(th2), s2 = __sinf(th2);
            for (int q = 0; q < NQ - 1; ++q) {
                apply_cnot(sre, sim_, q, q + 1, tid);
                apply_diag(sre, sim_, q + 1, c2,-s2, c2,s2, tid);          // RZ(0.2t)
                apply_cnot(sre, sim_, q, q + 1, tid);
            }
        }
    } break;
    }

    measure_all(sre, sim_, feats + (circ * BSZ + b) * FEATW, acc, tid);
}

// ---------------------------------------------------------------------
// Gates + LSTM update. One block = 16 batch rows; 6 waves, wave w does
// circuit w's feats(16x30) @ R[w](30x10) as one 16x16x32 f16 WMMA (K=30->32).
// grid 128, block 192.
// ---------------------------------------------------------------------
__global__ void gates_lstm_kernel(const float* __restrict__ feats,
                                  const float* __restrict__ R,
                                  float* __restrict__ h,
                                  float* __restrict__ c_state,
                                  float* __restrict__ out, int t) {
    __shared__ float g[6][16][16];
    const int tileM = blockIdx.x;
    const int wave  = threadIdx.x >> 5;     // circuit id 0..5
    const int lane  = threadIdx.x & 31;
    const int half  = lane >> 4;
    const int M     = tileM * 16 + (lane & 15);
    const int col16 = lane & 15;
    const int colC  = (col16 < NQ) ? col16 : 0;
    const float cm  = (col16 < NQ) ? 1.f : 0.f;

    __builtin_prefetch(feats + (wave * BSZ + M) * FEATW, 0, 0);

    v16h a, bm;
#pragma unroll
    for (int e = 0; e < 16; ++e) {
        const int v = e >> 1, hi = e & 1;
        const int k = (v < 4) ? (8 * half + 2 * v + hi)
                              : (16 + 8 * half + 2 * (v - 4) + hi);
        const int   kc = (k < FEATW) ? k : 0;
        const float km = (k < FEATW) ? 1.f : 0.f;
        const float av = feats[(wave * BSZ + M) * FEATW + kc] * km;
        const float bv = R[(wave * FEATW + kc) * NQ + colC] * (km * cm);
        a[e]  = (_Float16)av;
        bm[e] = (_Float16)bv;
    }
    v8f acc = {};
    acc = __builtin_amdgcn_wmma_f32_16x16x32_f16(false, a, false, bm,
                                                 (short)0, acc, false, false);
    const bool use_tanh = (wave == 3) || (wave == 5);
#pragma unroll
    for (int r = 0; r < 8; ++r) {
        const int row = r + 8 * half;
        float v = acc[r];
        v = use_tanh ? tanhf(v) : (1.f / (1.f + __expf(-v)));
        g[wave][row][col16] = v;
    }
    __syncthreads();

    const int idx = threadIdx.x;
    if (idx < 160) {                        // 16 rows x 10 cols
        const int row = idx / NQ, cc = idx % NQ;
        const int b = tileM * 16 + row;
        const float f  = g[0][row][cc];
        const float i_ = g[1][row][cc];
        const float o  = g[2][row][cc];
        const float gg = g[3][row][cc];
        const float mq = g[4][row][cc];
        const float tm = g[5][row][cc];
        const float cold = c_state[b * NQ + cc];
        const float cnew = (f * cold + i_ * gg) * mq;
        const float hnew = o * tanhf(cnew) + 0.1f * tm;
        c_state[b * NQ + cc] = cnew;
        h[b * NQ + cc] = hnew;
        out[(b * TT + t) * NQ + cc] = hnew;  // output is (B, T, N)
    }
}

// ---------------------------------------------------------------------
extern "C" void kernel_launch(void* const* d_in, const int* in_sizes, int n_in,
                              void* d_out, int out_size, void* d_ws, size_t ws_size,
                              hipStream_t stream) {
    const float* x    = (const float*)d_in[0];   // (B, T, N)
    const float* Wx   = (const float*)d_in[1];   // (N, 81)
    const float* Wh   = (const float*)d_in[2];   // (N, 81)
    const float* bias = (const float*)d_in[3];   // (81,)
    const float* R    = (const float*)d_in[4];   // (6, 30, N)
    float* out = (float*)d_out;                  // (B, T, N)

    float* ws     = (float*)d_ws;
    float* h      = ws;                          // B*N
    float* cst    = h + BSZ * NQ;                // B*N
    float* p      = cst + BSZ * NQ;              // B*81
    float* feats  = p + BSZ * PW;                // 6*B*30

    zero_kernel<<<(BSZ * NQ * 2 + 255) / 256, 256, 0, stream>>>(h, BSZ * NQ * 2);

    for (int t = 0; t < TT; ++t) {
        gemm_p_kernel<<<dim3(BSZ / 16, 6), 32, 0, stream>>>(x, t, Wx, Wh, bias, h, p);
        circuits_kernel<<<dim3(BSZ, 6), 256, 0, stream>>>(p, feats);
        gates_lstm_kernel<<<BSZ / 16, 192, 0, stream>>>(feats, R, h, cst, out, t);
    }
}
